// LSTM_26182120636716
// MI455X (gfx1250) — matrix operand order repaired and checked by reading, not compiled, not run on hardware
//
#include <hip/hip_runtime.h>
#include <cstdint>

// ---------------------------------------------------------------------------
// LSTM T=512 B=256 F=512 H=512 — persistent fused scan for MI455X (gfx1250).
//
// Variant A (ws >= ~142MB): x pre-swizzled to bf16 WMMA A-fragment layout,
//   h kept in A-fragment layout in a double-buffered global array written
//   directly by the elementwise stage. x fragments burst-loaded BEFORE the
//   grid-barrier spin (latency hidden by the wait); h fragments burst-loaded
//   after the acquire with a sched_barrier pinning all loads before the
//   first WMMA (partial loadcnt waits -> ~one L2 round trip exposed).
// Variant B (small ws): LDS staging; h tile copied with
//   global_load_async_to_lds_b128; same sched_barrier treatment.
// ---------------------------------------------------------------------------

typedef __attribute__((ext_vector_type(16))) __bf16 v16bf;
typedef __attribute__((ext_vector_type(8)))  float  v8f;

#define T_STEPS 512
#define BATCH   256
#define FDIM    512
#define HDIM    512
#define G4H     2048
#define LDSK    260              // padded LDS row stride in u32 (520 bf16)

// workspace layout (bytes)
#define OFF_WIH  0u
#define OFF_WHH  2097152u
#define OFF_BIAS 4194304u
#define OFF_PH   4202496u        // 2 x 256KB h buffers (zeroed by prep)
#define OFF_BAR  4726784u        // barrier counter (zeroed by prep)
#define ZERO_U32 131136u         // (512KB + 256B)/4
#define OFF_XF   8388608u        // 128MB x fragments (variant A only)
#define NEED_A   142606336ull

union AF { v16bf v; unsigned u[8]; uint4 q[2]; };
union CF { v8f   v; float    f[8]; };

static __device__ __forceinline__ unsigned bf16_rne(float f) {
  unsigned u = __float_as_uint(f);
  return (u + 0x7FFFu + ((u >> 16) & 1u)) >> 16;
}
static __device__ __forceinline__ unsigned pack2(float a, float b) {
  return bf16_rne(a) | (bf16_rne(b) << 16);
}
static __device__ __forceinline__ float sigm(float v) {
  return 1.0f / (1.0f + __expf(-v));
}

// ---------------------------------------------------------------------------
// Prep: pack W_ih/W_hh into WMMA B-fragment bf16 layout
//   packed[nt][kt][lane][j]: N = nt*16+(lane&15), K = kt*32+(j&8?16:0)+(lane&16?8:0)+(j&7)
// combined bias, zero h buffers + barrier.
// ---------------------------------------------------------------------------
__global__ void lstm_prep(const float* __restrict__ Wih, const float* __restrict__ Whh,
                          const float* __restrict__ bih, const float* __restrict__ bhh,
                          unsigned short* __restrict__ wih_p, unsigned short* __restrict__ whh_p,
                          float* __restrict__ bias, unsigned* __restrict__ zero_u32) {
  const long long PACK = (long long)G4H * HDIM;        // 1048576 per weight
  const long long total = 2 * PACK + G4H + ZERO_U32;
  long long i0 = (long long)blockIdx.x * blockDim.x + threadIdx.x;
  long long st = (long long)gridDim.x * blockDim.x;
  for (long long i = i0; i < total; i += st) {
    if (i < 2 * PACK) {
      const bool ih = (i < PACK);
      const float* W = ih ? Wih : Whh;
      unsigned short* P = ih ? wih_p : whh_p;
      unsigned p  = (unsigned)(ih ? i : i - PACK);
      unsigned j  = p & 15u;
      unsigned l  = (p >> 4) & 31u;
      unsigned kt = (p >> 9) & 15u;
      unsigned nt = p >> 13;
      unsigned N  = nt * 16u + (l & 15u);
      unsigned K  = kt * 32u + ((j & 8u) ? 16u : 0u) + ((l & 16u) ? 8u : 0u) + (j & 7u);
      P[p] = (unsigned short)bf16_rne(W[(size_t)N * HDIM + K]);
    } else if (i < 2 * PACK + G4H) {
      int j = (int)(i - 2 * PACK);
      bias[j] = bih[j] + bhh[j];
    } else {
      zero_u32[i - 2 * PACK - G4H] = 0u;
    }
  }
}

// ---------------------------------------------------------------------------
// Variant A prep2: swizzle x (T,B,F) f32 -> bf16 A-fragment layout
//   xf[(t*16+bt)*8192 + (kt*32+lane)*16 + j], lane=(row|lh<<4),
//   f = kt*32 + (j&8?16:0) + lh*8 + (j&7)
// ---------------------------------------------------------------------------
__global__ void lstm_xpack(const float* __restrict__ x, unsigned* __restrict__ xf_u32) {
  const long long total = (long long)T_STEPS * 16 * 8192 / 2;  // 33.5M u32
  long long i0 = (long long)blockIdx.x * blockDim.x + threadIdx.x;
  long long st = (long long)gridDim.x * blockDim.x;
  for (long long i = i0; i < total; i += st) {
    long long e = i * 2;
    unsigned tb = (unsigned)(e >> 13);       // t*16 + bt
    unsigned q  = (unsigned)(e & 8191);
    unsigned j  = q & 15u;
    unsigned l  = (q >> 4) & 31u;
    unsigned kt = q >> 9;
    unsigned row = l & 15u, lh = l >> 4;
    unsigned f  = kt * 32u + ((j & 8u) ? 16u : 0u) + lh * 8u + (j & 7u);
    unsigned t  = tb >> 4, bt = tb & 15u;
    const float* src = x + ((size_t)t * BATCH + bt * 16u + row) * FDIM + f;
    xf_u32[i] = pack2(src[0], src[1]);
  }
}

// ---------------------------------------------------------------------------
// Variant A scan: 512 blocks x 128 threads; zero LDS staging.
// ---------------------------------------------------------------------------
__global__ void __launch_bounds__(128, 1) lstm_scanA(
    const unsigned short* __restrict__ xf,
    const unsigned short* __restrict__ wih_p,
    const unsigned short* __restrict__ whh_p,
    const float* __restrict__ bias,
    unsigned short* __restrict__ ph,       // 2 x 16bt x 8192 A-fragment bf16
    unsigned* __restrict__ bar,
    float* __restrict__ out) {
  __shared__ float gbuf[4][16][16];
  __shared__ float cbuf[256];

  const int tid = threadIdx.x, lane = tid & 31, w = tid >> 5;
  const int bt = blockIdx.x >> 5, ht = blockIdx.x & 31;
  const int b0 = bt * 16, ntg = w * 32 + ht;
  const unsigned nblocks = gridDim.x;

  cbuf[tid * 2] = 0.0f; cbuf[tid * 2 + 1] = 0.0f;

  const unsigned short* bx_base = wih_p + (size_t)ntg * 8192;  // hoisted to regs
  const unsigned short* bh_base = whh_p + (size_t)ntg * 8192;
  const float bsc = bias[w * HDIM + ht * 16 + (lane & 15)];    // per-column bias

  for (int t = 0; t < T_STEPS; ++t) {
    // ---- burst x fragments first: latency hidden behind the barrier spin ----
    const unsigned short* xfb = xf + (((size_t)t * 16 + bt) << 13);
    AF ax[16];
    #pragma unroll
    for (int kt = 0; kt < 16; ++kt) {
      const uint4* pax = (const uint4*)(xfb + (kt * 32 + lane) * 16);
      ax[kt].q[0] = pax[0]; ax[kt].q[1] = pax[1];
    }
    __builtin_amdgcn_sched_barrier(0);   // keep the burst above the spin

    // wait: all blocks finished writing h(t) fragments (t=0 trivially true)
    if (tid == 0) {
      unsigned target = nblocks * (unsigned)t;
      while (__hip_atomic_load(bar, __ATOMIC_ACQUIRE, __HIP_MEMORY_SCOPE_AGENT) < target)
        __builtin_amdgcn_s_sleep(1);
    }
    __syncthreads();
    __threadfence();                     // acquire other blocks' h stores

    // ---- burst all 32 h-fragment b128s, pinned before the WMMAs ----
    const unsigned short* phb = ph + (size_t)(t & 1) * 131072 + ((size_t)bt << 13);
    AF ah[16];
    #pragma unroll
    for (int kt = 0; kt < 16; ++kt) {
      const uint4* pah = (const uint4*)(phb + (kt * 32 + lane) * 16);
      ah[kt].q[0] = pah[0]; ah[kt].q[1] = pah[1];
    }
    __builtin_amdgcn_sched_barrier(0);   // all loads in flight before 1st wmma

    // ---- 32 WMMAs, two accumulator chains; bias pre-loaded into accX ----
    CF accX, accH;
    #pragma unroll
    for (int r = 0; r < 8; ++r) { accX.f[r] = bsc; accH.f[r] = 0.0f; }
    #pragma unroll
    for (int kt = 0; kt < 16; ++kt) {
      v16bf bx = *(const v16bf*)(bx_base + kt * 512 + lane * 16);
      v16bf bh = *(const v16bf*)(bh_base + kt * 512 + lane * 16);
      accX.v = __builtin_amdgcn_wmma_f32_16x16x32_bf16(false, ax[kt].v, false, bx,
                                                       (short)0, accX.v, false, false);
      accH.v = __builtin_amdgcn_wmma_f32_16x16x32_bf16(false, ah[kt].v, false, bh,
                                                       (short)0, accH.v, false, false);
    }

    {
      const int r0 = (lane & 16) ? 8 : 0, col = lane & 15;
      #pragma unroll
      for (int r = 0; r < 8; ++r) gbuf[w][r0 + r][col] = accX.f[r] + accH.f[r];
    }
    __syncthreads();

    // ---- elementwise LSTM cell (2 adjacent cols per thread) ----
    {
      const int e0 = tid * 2, row = e0 >> 4, col = e0 & 15;
      const int hc = ht * 16 + col;
      float cn0 = sigm(gbuf[1][row][col])   * cbuf[e0]
                + sigm(gbuf[0][row][col])   * tanhf(gbuf[2][row][col]);
      float cn1 = sigm(gbuf[1][row][col+1]) * cbuf[e0+1]
                + sigm(gbuf[0][row][col+1]) * tanhf(gbuf[2][row][col+1]);
      float hn0 = sigm(gbuf[3][row][col])   * tanhf(cn0);
      float hn1 = sigm(gbuf[3][row][col+1]) * tanhf(cn1);
      cbuf[e0] = cn0; cbuf[e0 + 1] = cn1;

      size_t ob = (size_t)(b0 + row) * (T_STEPS * HDIM) + (size_t)t * HDIM + hc;
      float2 hv; hv.x = hn0; hv.y = hn1;
      *(float2*)(out + ob) = hv;

      // h(t+1) straight into the A-fragment layout of the other buffer
      unsigned* phw = (unsigned*)(ph + (size_t)((t + 1) & 1) * 131072);
      int kp = hc & 31, ktw = hc >> 5;
      int lh = (kp >> 3) & 1;
      int jj = (kp & 7) | ((kp & 16) >> 1);
      int lw = row + lh * 16;
      phw[bt * 4096 + ktw * 256 + lw * 8 + (jj >> 1)] = pack2(hn0, hn1);

      if (t == T_STEPS - 1) {
        size_t base_h = (size_t)BATCH * T_STEPS * HDIM;
        size_t base_c = base_h + (size_t)BATCH * HDIM;
        size_t oo = (size_t)(b0 + row) * HDIM + hc;
        float2 cv; cv.x = cn0; cv.y = cn1;
        *(float2*)(out + base_h + oo) = hv;
        *(float2*)(out + base_c + oo) = cv;
      }
    }

    __threadfence();                     // release h stores
    __syncthreads();
    if (tid == 0)
      __hip_atomic_fetch_add(bar, 1u, __ATOMIC_RELEASE, __HIP_MEMORY_SCOPE_AGENT);
  }
}

// ---------------------------------------------------------------------------
// Variant B scan (small ws): LDS staging; h row-major double-buffered in ph;
// h tile staged with async global->LDS copies.
// ---------------------------------------------------------------------------
__global__ void __launch_bounds__(128, 1) lstm_scanB(
    const float* __restrict__ x,
    const unsigned short* __restrict__ wih_p,
    const unsigned short* __restrict__ whh_p,
    const float* __restrict__ bias,
    unsigned short* __restrict__ hb,       // 2 x (256x512) row-major bf16
    unsigned* __restrict__ bar,
    float* __restrict__ out) {
  __shared__ unsigned xs[16 * LDSK];
  __shared__ unsigned hs[16 * LDSK];
  __shared__ float gbuf[4][16][16];
  __shared__ float cbuf[256];

  const int tid = threadIdx.x, lane = tid & 31, w = tid >> 5;
  const int bt = blockIdx.x >> 5, ht = blockIdx.x & 31;
  const int b0 = bt * 16, ntg = w * 32 + ht;
  const unsigned nblocks = gridDim.x;

  cbuf[tid * 2] = 0.0f; cbuf[tid * 2 + 1] = 0.0f;

  const unsigned short* bx_base = wih_p + (size_t)ntg * 8192;
  const unsigned short* bh_base = whh_p + (size_t)ntg * 8192;
  const float bsc = bias[w * HDIM + ht * 16 + (lane & 15)];

  for (int t = 0; t < T_STEPS; ++t) {
    // ---- stage x tile (f32 -> bf16) before the spin: latency hidden ----
    // (xs was last read before the gbuf barrier of step t-1: safe to overwrite)
    {
      const float4* xsrc = (const float4*)(x + ((size_t)t * BATCH + b0) * FDIM);
      #pragma unroll
      for (int j = 0; j < 16; ++j) {
        int q = j * 128 + tid;
        float4 v = xsrc[q];
        int row = q >> 7, k = (q & 127) * 4;
        xs[row * LDSK + (k >> 1)]     = pack2(v.x, v.y);
        xs[row * LDSK + (k >> 1) + 1] = pack2(v.z, v.w);
      }
      if (t + 1 < T_STEPS) {
        const float* xn = x + ((size_t)(t + 1) * BATCH + b0) * FDIM;
        __builtin_prefetch(xn + tid * 64, 0, 1);
      }
    }
    __builtin_amdgcn_sched_barrier(0);

    if (tid == 0) {
      unsigned target = nblocks * (unsigned)t;
      while (__hip_atomic_load(bar, __ATOMIC_ACQUIRE, __HIP_MEMORY_SCOPE_AGENT) < target)
        __builtin_amdgcn_s_sleep(1);
    }
    __syncthreads();
    __threadfence();

    // ---- async-copy h tile (bf16, 16KB) global -> LDS (padded rows) ----
    {
      const unsigned short* hsrc = hb + (size_t)(t & 1) * 131072 + (size_t)b0 * HDIM;
      unsigned hs0 = (unsigned)(size_t)&hs[0];   // generic LDS ptr: low32 = LDS offset
      #pragma unroll
      for (int j = 0; j < 8; ++j) {
        int q = j * 128 + tid;                   // 1024 x 16B chunks
        unsigned ldst = hs0 + (unsigned)(q >> 6) * 1040u + (unsigned)(q & 63) * 16u;
        unsigned long long ga = (unsigned long long)(const void*)hsrc
                              + (unsigned long long)q * 16ull;
        asm volatile("global_load_async_to_lds_b128 %0, %1, off"
                     :: "v"(ldst), "v"(ga) : "memory");
      }
    }
    asm volatile("s_wait_asynccnt 0x0" ::: "memory");
    __syncthreads();

    // ---- preload all LDS A-fragments into registers, then WMMA ----
    const int arow = lane & 15;
    const int koff = (lane & 16) ? 8 : 0;
    AF ax[16], ah[16];
    #pragma unroll
    for (int kt = 0; kt < 16; ++kt) {
      const int kb = kt * 32 + koff;
      #pragma unroll
      for (int m = 0; m < 4; ++m) {
        ax[kt].u[m]     = xs[arow * LDSK + ((kb + 2 * m) >> 1)];
        ax[kt].u[m + 4] = xs[arow * LDSK + ((kb + 16 + 2 * m) >> 1)];
        ah[kt].u[m]     = hs[arow * LDSK + ((kb + 2 * m) >> 1)];
        ah[kt].u[m + 4] = hs[arow * LDSK + ((kb + 16 + 2 * m) >> 1)];
      }
    }
    __builtin_amdgcn_sched_barrier(0);   // all ds loads issued before 1st wmma

    CF accX, accH;
    #pragma unroll
    for (int r = 0; r < 8; ++r) { accX.f[r] = bsc; accH.f[r] = 0.0f; }
    #pragma unroll
    for (int kt = 0; kt < 16; ++kt) {
      v16bf bx = *(const v16bf*)(bx_base + kt * 512 + lane * 16);
      v16bf bh = *(const v16bf*)(bh_base + kt * 512 + lane * 16);
      accX.v = __builtin_amdgcn_wmma_f32_16x16x32_bf16(false, ax[kt].v, false, bx,
                                                       (short)0, accX.v, false, false);
      accH.v = __builtin_amdgcn_wmma_f32_16x16x32_bf16(false, ah[kt].v, false, bh,
                                                       (short)0, accH.v, false, false);
    }
    {
      const int r0 = (lane & 16) ? 8 : 0, col = lane & 15;
      #pragma unroll
      for (int r = 0; r < 8; ++r) gbuf[w][r0 + r][col] = accX.f[r] + accH.f[r];
    }
    __syncthreads();

    {
      const int e0 = tid * 2, row = e0 >> 4, col = e0 & 15;
      const int hc = ht * 16 + col;
      float cn0 = sigm(gbuf[1][row][col])   * cbuf[e0]
                + sigm(gbuf[0][row][col])   * tanhf(gbuf[2][row][col]);
      float cn1 = sigm(gbuf[1][row][col+1]) * cbuf[e0+1]
                + sigm(gbuf[0][row][col+1]) * tanhf(gbuf[2][row][col+1]);
      float hn0 = sigm(gbuf[3][row][col])   * tanhf(cn0);
      float hn1 = sigm(gbuf[3][row][col+1]) * tanhf(cn1);
      cbuf[e0] = cn0; cbuf[e0 + 1] = cn1;

      size_t ob = (size_t)(b0 + row) * (T_STEPS * HDIM) + (size_t)t * HDIM + hc;
      float2 hv; hv.x = hn0; hv.y = hn1;
      *(float2*)(out + ob) = hv;

      unsigned* hbw = (unsigned*)(hb + (size_t)((t + 1) & 1) * 131072);
      hbw[((size_t)(b0 + row) * HDIM + hc) >> 1] = pack2(hn0, hn1);

      if (t == T_STEPS - 1) {
        size_t base_h = (size_t)BATCH * T_STEPS * HDIM;
        size_t base_c = base_h + (size_t)BATCH * HDIM;
        size_t oo = (size_t)(b0 + row) * HDIM + hc;
        float2 cv; cv.x = cn0; cv.y = cn1;
        *(float2*)(out + base_h + oo) = hv;
        *(float2*)(out + base_c + oo) = cv;
      }
    }

    __threadfence();
    __syncthreads();
    if (tid == 0)
      __hip_atomic_fetch_add(bar, 1u, __ATOMIC_RELEASE, __HIP_MEMORY_SCOPE_AGENT);
  }
}

// ---------------------------------------------------------------------------
extern "C" void kernel_launch(void* const* d_in, const int* in_sizes, int n_in,
                              void* d_out, int out_size, void* d_ws, size_t ws_size,
                              hipStream_t stream) {
  const float* x   = (const float*)d_in[0];
  const float* wih = (const float*)d_in[1];
  const float* whh = (const float*)d_in[2];
  const float* bih = (const float*)d_in[3];
  const float* bhh = (const float*)d_in[4];

  uint8_t* ws = (uint8_t*)d_ws;
  unsigned short* wih_p = (unsigned short*)(ws + OFF_WIH);
  unsigned short* whh_p = (unsigned short*)(ws + OFF_WHH);
  float*          bias  = (float*)(ws + OFF_BIAS);
  unsigned short* ph    = (unsigned short*)(ws + OFF_PH);
  unsigned*       bar   = (unsigned*)(ws + OFF_BAR);
  unsigned short* xf    = (unsigned short*)(ws + OFF_XF);
  float*          out   = (float*)d_out;

  lstm_prep<<<2048, 512, 0, stream>>>(wih, whh, bih, bhh, wih_p, whh_p, bias,
                                      (unsigned*)ph /* zero region start */);

  if (ws_size >= (size_t)NEED_A) {
    lstm_xpack<<<16384, 512, 0, stream>>>(x, (unsigned*)xf);
    lstm_scanA<<<512, 128, 0, stream>>>(xf, wih_p, whh_p, bias, ph, bar, out);
  } else {
    lstm_scanB<<<512, 128, 0, stream>>>(x, wih_p, whh_p, bias, ph, bar, out);
  }
}
